// GraphAttentionNetwork_39487929319659
// MI455X (gfx1250) — matrix operand order repaired and checked by reading, not compiled, not run on hardware
//
#include <hip/hip_runtime.h>
#include <hip/hip_bf16.h>
#include <cstdint>

#define NN 40000
#define NE 640000
#define HIDD 128
#define E2T (NE + NN)

typedef __attribute__((ext_vector_type(16))) __bf16 bf16x16;
typedef __attribute__((ext_vector_type(8)))  __bf16 bf16x8;
typedef __attribute__((ext_vector_type(8)))  float  f32x8;

// ---------------------------------------------------------------------------
// Row-major GEMM  C[M,Nc] = A[M,K] @ B[K,Nc] (+bias), f32 in/out, bf16 WMMA
// compute (v_wmma_f32_16x16x32_bf16). Compile-time K and NT (Nc = 16*NT) so
// the accumulator array stays in VGPRs (no scratch spills) and all fragment
// gathers are aligned ds_load_b128.
// Block = 256 thr = 8 waves; block covers 128 rows (wave w -> rows base+16w).
// ---------------------------------------------------------------------------
template <int K, int NT>
__global__ __launch_bounds__(256) void k_gemm_bf16(
    const float* __restrict__ A, const float* __restrict__ B,
    const float* __restrict__ bias, float* __restrict__ C, int M)
{
  constexpr int Nc = NT * 16;
  // B staged in WMMA fragment order: slot = ((t*32 + lane)<<4) + i,
  // holding B[k0 + (lane<16?0:16) + i][t*16 + (lane&15)].
  __shared__ __attribute__((aligned(32))) __bf16 sB[32 * Nc];
  __shared__ __attribute__((aligned(16))) __bf16 sA[8][16 * 32];   // per-wave A tile

  const int tid  = threadIdx.x;
  const int wave = tid >> 5;
  const int lane = tid & 31;
  const int row0 = blockIdx.x * 128 + wave * 16;

  f32x8 acc[NT];
#pragma unroll
  for (int t = 0; t < NT; ++t)
#pragma unroll
    for (int j = 0; j < 8; ++j) acc[t][j] = 0.0f;

#pragma unroll
  for (int k0 = 0; k0 < K; k0 += 32) {
    // ---- cooperative B staging (f32 -> bf16), fragment-order swizzle ----
    for (int i = tid; i < 32 * Nc; i += 256) {
      int kk = i / Nc, nn = i - kk * Nc;
      int t      = nn >> 4;
      int lane_t = (nn & 15) | (kk & 16);
      int slot   = ((t * 32 + lane_t) << 4) + (kk & 15);
      sB[slot] = (__bf16)B[(size_t)(k0 + kk) * Nc + nn];
    }
    // ---- per-wave A staging (coalesced: lane = column), zero-pad past M ----
    {
      int kk = lane;
#pragma unroll
      for (int r = 0; r < 16; ++r) {
        int gr = row0 + r;
        float v = (gr < M) ? A[(size_t)gr * K + k0 + kk] : 0.0f;
        sA[wave][r * 32 + kk] = (__bf16)v;
        if (k0 + 32 < K && gr < M)
          __builtin_prefetch(&A[(size_t)gr * K + k0 + 32 + kk], 0, 0);
      }
    }
    __syncthreads();

    // A fragment: lane m=l%16; elements {0..15} at K kb..kb+15 (kb = 16*(l/16))
    const int m  = lane & 15;
    const int kb = (lane < 16) ? 0 : 16;
    bf16x8 alo = *(const bf16x8*)&sA[wave][m * 32 + kb];
    bf16x8 ahi = *(const bf16x8*)&sA[wave][m * 32 + 8 + kb];
    bf16x16 af = __builtin_shufflevector(alo, ahi, 0, 1, 2, 3, 4, 5, 6, 7,
                                         8, 9, 10, 11, 12, 13, 14, 15);
#pragma unroll
    for (int t = 0; t < NT; ++t) {
      bf16x16 bfr = *(const bf16x16*)&sB[(t * 32 + lane) << 4];
      acc[t] = __builtin_amdgcn_wmma_f32_16x16x32_bf16(
          false, af, false, bfr, (short)0, acc[t], false, false);
    }
    __syncthreads();
  }

  // store: VGPR r -> M = row0 + r (+8 for lanes 16..31); N = l%16 + 16t
  const int mofs = (lane < 16) ? 0 : 8;
#pragma unroll
  for (int t = 0; t < NT; ++t) {
    int n = t * 16 + (lane & 15);
    float bv = bias ? bias[n] : 0.0f;
#pragma unroll
    for (int r = 0; r < 8; ++r) {
      int gm = row0 + mofs + r;
      if (gm < M) C[(size_t)gm * Nc + n] = acc[t][r] + bv;
    }
  }
}

// ---------------------------------------------------------------------------
// Helpers / edge-phase kernels
// ---------------------------------------------------------------------------
__device__ __forceinline__ void edge_nodes(const int* __restrict__ ei,
                                           int e2, int& s, int& d) {
  if (e2 < NE) { s = ei[e2]; d = ei[NE + e2]; }
  else { s = e2 - NE; d = e2 - NE; }            // appended self-loops
}

__device__ __forceinline__ void atomicMaxF(float* addr, float val) {
  if (val >= 0.0f) atomicMax((int*)addr, __float_as_int(val));
  else             atomicMin((unsigned int*)addr, __float_as_uint(val));
}

// fold We[128,H*C] with att_edge[H,C] -> V[128,16] (cols >= H zeroed)
__global__ void k_fold_att(const float* __restrict__ We,
                           const float* __restrict__ att_e,
                           float* __restrict__ V, int H, int C) {
  int idx = blockIdx.x * blockDim.x + threadIdx.x;
  if (idx >= 128 * 16) return;
  int k = idx >> 4, h = idx & 15;
  float a = 0.0f;
  if (h < H)
    for (int c = 0; c < C; ++c) a += We[(size_t)k * HIDD + h * C + c] * att_e[h * C + c];
  V[idx] = a;
}

// zero self-loop region of ea_full and cnt
__global__ void k_loop_zero(float* __restrict__ ea_full, float* __restrict__ cnt) {
  int idx = blockIdx.x * blockDim.x + threadIdx.x;
  if (idx < NN * HIDD) ea_full[(size_t)NE * HIDD + idx] = 0.0f;
  if (idx < NN) cnt[idx] = 0.0f;
}

// accumulate incoming edge features per destination (fill_value='mean' prep)
__global__ void k_loop_accum(const int* __restrict__ ei,
                             float* __restrict__ ea_full, float* __restrict__ cnt) {
  long idx = (long)blockIdx.x * blockDim.x + threadIdx.x;
  if (idx >= (long)NE * HIDD) return;
  int e = (int)(idx >> 7), c = (int)(idx & 127);
  int d = ei[NE + e];
  atomicAdd(&ea_full[((size_t)NE + d) * HIDD + c], ea_full[(size_t)e * HIDD + c]);
  if (c == 0) atomicAdd(&cnt[d], 1.0f);
}

__global__ void k_loop_div(float* __restrict__ ea_full, const float* __restrict__ cnt) {
  int idx = blockIdx.x * blockDim.x + threadIdx.x;
  if (idx >= NN * HIDD) return;
  int n = idx >> 7;
  ea_full[(size_t)NE * HIDD + idx] /= fmaxf(cnt[n], 1.0f);
}

// a_s[n,h] = sum_c h[n,h*C+c]*att_src[h,c]; same for a_d
__global__ void k_att_scores(const float* __restrict__ hbuf,
                             const float* __restrict__ att_s,
                             const float* __restrict__ att_d,
                             float* __restrict__ a_s, float* __restrict__ a_d,
                             int H, int C) {
  int idx = blockIdx.x * blockDim.x + threadIdx.x;
  if (idx >= NN * H) return;
  int n = idx / H, h = idx - n * H;
  float ss = 0.0f, dd = 0.0f;
  for (int c = 0; c < C; ++c) {
    float v = hbuf[(size_t)n * HIDD + h * C + c];
    ss += v * att_s[h * C + c];
    dd += v * att_d[h * C + c];
  }
  a_s[idx] = ss; a_d[idx] = dd;
}

// zero agg / denom / bn sums, init m to -inf
__global__ void k_layer_init(float* __restrict__ agg, float* __restrict__ m,
                             float* __restrict__ denom, float* __restrict__ bns,
                             int H) {
  int idx = blockIdx.x * blockDim.x + threadIdx.x;
  if (idx < NN * HIDD) agg[idx] = 0.0f;
  if (idx < NN * H) { m[idx] = -3.0e38f; denom[idx] = 0.0f; }
  if (idx < 256) bns[idx] = 0.0f;
}

// alpha = leakyrelu(a_s[src]+a_d[dst]+a_e), segment max into m[dst]
__global__ void k_alpha(const int* __restrict__ ei,
                        const float* __restrict__ a_s, const float* __restrict__ a_d,
                        const float* __restrict__ a_e,
                        float* __restrict__ alpha, float* __restrict__ m, int H) {
  long idx = (long)blockIdx.x * blockDim.x + threadIdx.x;
  if (idx >= (long)E2T * H) return;
  int e2 = (int)(idx / H), h = (int)(idx - (long)e2 * H);
  int s, d; edge_nodes(ei, e2, s, d);
  float a = a_s[s * H + h] + a_d[d * H + h] + a_e[(size_t)e2 * 16 + h];
  a = (a > 0.0f) ? a : 0.2f * a;
  alpha[idx] = a;
  atomicMaxF(&m[d * H + h], a);
}

// expa = exp(alpha - m[dst]); segment sum into denom[dst]
__global__ void k_expsum(const int* __restrict__ ei,
                         float* __restrict__ alpha, const float* __restrict__ m,
                         float* __restrict__ denom, int H) {
  long idx = (long)blockIdx.x * blockDim.x + threadIdx.x;
  if (idx >= (long)E2T * H) return;
  int e2 = (int)(idx / H), h = (int)(idx - (long)e2 * H);
  int s, d; edge_nodes(ei, e2, s, d); (void)s;
  float ev = __expf(alpha[idx] - m[d * H + h]);
  alpha[idx] = ev;
  atomicAdd(&denom[d * H + h], ev);
}

// agg[dst] += (expa/denom[dst]) * h[src]
__global__ void k_scatter(const int* __restrict__ ei,
                          const float* __restrict__ alpha, const float* __restrict__ denom,
                          const float* __restrict__ hbuf, float* __restrict__ agg,
                          int H, int C) {
  long idx = (long)blockIdx.x * blockDim.x + threadIdx.x;
  if (idx >= (long)E2T * HIDD) return;
  int e2 = (int)(idx >> 7), c = (int)(idx & 127);
  int s, d; edge_nodes(ei, e2, s, d);
  int h = c / C;
  float w = alpha[(long)e2 * H + h] / (denom[d * H + h] + 1e-16f);
  atomicAdd(&agg[(size_t)d * HIDD + c], w * hbuf[(size_t)s * HIDD + c]);
}

// per-channel sum / sumsq over nodes (bias cancels inside training-mode BN)
__global__ __launch_bounds__(256) void k_bn_stats(const float* __restrict__ agg,
                                                  float* __restrict__ bns) {
  __shared__ float s1[256], s2[256];
  int tid = threadIdx.x;
  float a = 0.0f, b = 0.0f;
  for (long i = (long)blockIdx.x * 256 + tid; i < (long)NN * HIDD;
       i += (long)gridDim.x * 256) {                 // stride % 128 == 0 -> fixed channel
    float v = agg[i]; a += v; b += v * v;
  }
  s1[tid] = a; s2[tid] = b;
  __syncthreads();
  if (tid < 128) {
    atomicAdd(&bns[tid],       s1[tid] + s1[tid + 128]);
    atomicAdd(&bns[128 + tid], s2[tid] + s2[tid + 128]);
  }
}

// out = maybe_relu((agg - mu)*rsqrt(var+eps)*gamma + beta) + res
__global__ void k_bn_apply(const float* __restrict__ agg, const float* __restrict__ bns,
                           const float* __restrict__ gamma, const float* __restrict__ beta,
                           const float* __restrict__ res, float* __restrict__ out,
                           int do_relu) {
  long idx = (long)blockIdx.x * blockDim.x + threadIdx.x;
  if (idx >= (long)NN * HIDD) return;
  int c = (int)(idx & 127);
  float mu  = bns[c] * (1.0f / NN);
  float var = bns[128 + c] * (1.0f / NN) - mu * mu;
  float v = (agg[idx] - mu) * rsqrtf(var + 1e-5f) * gamma[c] + beta[c];
  if (do_relu) v = fmaxf(v, 0.0f);
  out[idx] = v + res[idx];
}

// ---------------------------------------------------------------------------
extern "C" void kernel_launch(void* const* d_in, const int* in_sizes, int n_in,
                              void* d_out, int out_size, void* d_ws, size_t ws_size,
                              hipStream_t stream) {
  const float* x      = (const float*)d_in[0];   // [N,64]
  const int*   ei     = (const int*)  d_in[1];   // [2,E]
  const float* eattr  = (const float*)d_in[2];   // [E,32]
  const float* node_W = (const float*)d_in[3];   // [64,128]
  const float* node_b = (const float*)d_in[4];
  const float* edge_W = (const float*)d_in[5];   // [32,128]
  const float* edge_b = (const float*)d_in[6];
  // per layer i (base 7 + 8*i): W, We, att_src, att_dst, att_edge, bias, gamma, beta

  // ---- workspace carve (floats) ----
  float* p = (float*)d_ws;
  float* ea_full = p;  p += (size_t)E2T * HIDD;   // [E+N,128]
  float* xA      = p;  p += (size_t)NN * HIDD;
  float* xB      = p;  p += (size_t)NN * HIDD;
  float* hbuf    = p;  p += (size_t)NN * HIDD;
  float* agg     = p;  p += (size_t)NN * HIDD;
  float* a_e     = p;  p += (size_t)E2T * 16;     // [E2,16] (H<=8 used)
  float* alpha   = p;  p += (size_t)E2T * 8;
  float* a_s     = p;  p += (size_t)NN * 8;
  float* a_d     = p;  p += (size_t)NN * 8;
  float* mbuf    = p;  p += (size_t)NN * 8;
  float* denom   = p;  p += (size_t)NN * 8;
  float* cnt     = p;  p += (size_t)NN;
  float* V_e     = p;  p += 128 * 16;
  float* bns     = p;  p += 256;

  auto cdiv = [](long a, long b) { return (unsigned)((a + b - 1) / b); };

  // input projections (WMMA, bf16 compute / f32 accumulate)
  k_gemm_bf16<64, 8><<<cdiv(NN, 128), 256, 0, stream>>>(x, node_W, node_b, xA, NN);
  k_gemm_bf16<32, 8><<<cdiv(NE, 128), 256, 0, stream>>>(eattr, edge_W, edge_b, ea_full, NE);

  // self-loop edge features = mean of incoming edge features per destination
  k_loop_zero <<<cdiv((long)NN * HIDD, 256), 256, 0, stream>>>(ea_full, cnt);
  k_loop_accum<<<cdiv((long)NE * HIDD, 256), 256, 0, stream>>>(ei, ea_full, cnt);
  k_loop_div  <<<cdiv((long)NN * HIDD, 256), 256, 0, stream>>>(ea_full, cnt);

  const int Hs[3] = {8, 8, 1};
  const int Cs[3] = {16, 16, 128};
  float* xin = xA;
  for (int i = 0; i < 3; ++i) {
    const float* W     = (const float*)d_in[7 + 8 * i + 0];
    const float* We    = (const float*)d_in[7 + 8 * i + 1];
    const float* att_s = (const float*)d_in[7 + 8 * i + 2];
    const float* att_d = (const float*)d_in[7 + 8 * i + 3];
    const float* att_e = (const float*)d_in[7 + 8 * i + 4];
    const float* gamma = (const float*)d_in[7 + 8 * i + 6];
    const float* beta  = (const float*)d_in[7 + 8 * i + 7];
    int H = Hs[i], C = Cs[i];
    float* xout = (i == 0) ? xB : (i == 1) ? xA : (float*)d_out;

    // fold edge weights with att_edge; a_e = ea_full @ V_e   (WMMA, Nc=16)
    k_fold_att<<<cdiv(128 * 16, 256), 256, 0, stream>>>(We, att_e, V_e, H, C);
    k_gemm_bf16<128, 1><<<cdiv(E2T, 128), 256, 0, stream>>>(ea_full, V_e, nullptr, a_e, E2T);

    // h = x @ W (WMMA); per-node attention scores
    k_gemm_bf16<128, 8><<<cdiv(NN, 128), 256, 0, stream>>>(xin, W, nullptr, hbuf, NN);
    k_att_scores<<<cdiv((long)NN * H, 256), 256, 0, stream>>>(hbuf, att_s, att_d,
                                                              a_s, a_d, H, C);

    // edge phase: leakyrelu -> segment max -> exp/segment-sum -> weighted scatter
    k_layer_init<<<cdiv((long)NN * HIDD, 256), 256, 0, stream>>>(agg, mbuf, denom, bns, H);
    k_alpha  <<<cdiv((long)E2T * H, 256), 256, 0, stream>>>(ei, a_s, a_d, a_e, alpha, mbuf, H);
    k_expsum <<<cdiv((long)E2T * H, 256), 256, 0, stream>>>(ei, alpha, mbuf, denom, H);
    k_scatter<<<cdiv((long)E2T * HIDD, 256), 256, 0, stream>>>(ei, alpha, denom, hbuf, agg, H, C);

    // BatchNorm (training stats; bias cancels) + relu (not last) + residual
    k_bn_stats<<<512, 256, 0, stream>>>(agg, bns);
    k_bn_apply<<<cdiv((long)NN * HIDD, 256), 256, 0, stream>>>(agg, bns, gamma, beta,
                                                               xin, xout, i < 2 ? 1 : 0);
    xin = xout;
  }
}
